// MHA_6081673691665
// MI455X (gfx1250) — matrix operand order repaired and checked by reading, not compiled, not run on hardware
//
#include <hip/hip_runtime.h>

typedef __bf16 bf16_t;
typedef __attribute__((ext_vector_type(16))) __bf16 v16bf;
typedef __attribute__((ext_vector_type(8)))  __bf16 v8bf;
typedef __attribute__((ext_vector_type(8)))  float  v8f;

#define WMMA_BF16(a, b, c) \
  __builtin_amdgcn_wmma_f32_16x16x32_bf16(false, (a), false, (b), (short)0, (c), false, false)

// Hard scheduling fence: nothing may be reordered across this point.
// Used to pin "all fragment ds_loads" before "all WMMAs" so the waitcnt
// inserter emits staggered s_wait_dscnt instead of per-WMMA dscnt-0 stalls.
#if defined(__has_builtin)
#  if __has_builtin(__builtin_amdgcn_sched_barrier)
#    define SCHED_FENCE() __builtin_amdgcn_sched_barrier(0)
#  endif
#endif
#ifndef SCHED_FENCE
#  define SCHED_FENCE()
#endif

namespace {
constexpr int kB  = 2;
constexpr int kSq = 2048;
constexpr int kSk = 2048;
constexpr int kH  = 16;
constexpr int kD  = 128;

constexpr int BR_WAVE = 16;                // Q rows per wave
constexpr int NWAVES  = 4;
constexpr int BR      = BR_WAVE * NWAVES;  // 64 Q rows per block
constexpr int BC      = 32;                // KV cols per iteration

// LDS strides in bf16 elements; all row strides are multiples of 8 elements
// (16 bytes) so every v8bf fragment load is a 16B-aligned ds_load_b128.
constexpr int KSTR = kD + 8;   // 136 el = 272 B per K row
constexpr int VSTR = BC + 8;   // 40 el  = 80 B per Vt row
constexpr int PSTR = BC + 8;   // 40 el  = 80 B per P row
}

__global__ __launch_bounds__(NWAVES * 32) void fa_fwd_bf16_wmma(
    const float* __restrict__ q, const float* __restrict__ kv,
    float* __restrict__ out)
{
  __shared__ __align__(32) bf16_t sK [BC * KSTR];               //  8704 B, row major
  __shared__ __align__(32) bf16_t sVt[kD * VSTR];               // 10240 B, transposed
  __shared__ __align__(32) bf16_t sP [NWAVES * BR_WAVE * PSTR]; //  5120 B

  const int tid  = threadIdx.x;
  const int lane = tid & 31;
  const int wave = tid >> 5;
  const int half = lane >> 4;    // wave32 half: selects K-group in fragments
  const int l16  = lane & 15;

  const int qt = blockIdx.x;
  const int h  = blockIdx.y;
  const int b  = blockIdx.z;

  const int q_base = qt * BR;
  const int a_row  = q_base + wave * BR_WAVE + l16;  // this lane's A-fragment row

  // ---- Q fragments: fp32 (float4 loads) -> bf16, directly in WMMA A layout ----
  // element e of chunk c maps to head-dim k = c*32 + (e/8)*16 + half*8 + e%8
  v16bf aq[4];
  {
    const float* qrow = q + (((size_t)b * kSq + a_row) * kH + h) * kD;
    #pragma unroll
    for (int c = 0; c < 4; ++c) {
      const int k0 = c * 32 + half * 8;
      const float4 f0 = *(const float4*)(qrow + k0);
      const float4 f1 = *(const float4*)(qrow + k0 + 4);
      const float4 f2 = *(const float4*)(qrow + k0 + 16);
      const float4 f3 = *(const float4*)(qrow + k0 + 20);
      aq[c][0]  = (bf16_t)f0.x; aq[c][1]  = (bf16_t)f0.y;
      aq[c][2]  = (bf16_t)f0.z; aq[c][3]  = (bf16_t)f0.w;
      aq[c][4]  = (bf16_t)f1.x; aq[c][5]  = (bf16_t)f1.y;
      aq[c][6]  = (bf16_t)f1.z; aq[c][7]  = (bf16_t)f1.w;
      aq[c][8]  = (bf16_t)f2.x; aq[c][9]  = (bf16_t)f2.y;
      aq[c][10] = (bf16_t)f2.z; aq[c][11] = (bf16_t)f2.w;
      aq[c][12] = (bf16_t)f3.x; aq[c][13] = (bf16_t)f3.y;
      aq[c][14] = (bf16_t)f3.z; aq[c][15] = (bf16_t)f3.w;
    }
  }

  const v8f vzero = {};
  v8f o[8];                      // O accum: 16 x 128 per wave (8 N-chunks)
  #pragma unroll
  for (int nc = 0; nc < 8; ++nc) o[nc] = vzero;
  float m_i[8], l_i[8];          // per-row running max / sum (row = i + 8*half)
  #pragma unroll
  for (int i = 0; i < 8; ++i) { m_i[i] = -1.0e30f; l_i[i] = 0.0f; }

  const float softmax_scale = 0.08838834764831845f;  // 1/sqrt(128)

  const int n_tiles = (q_base + BR) / BC;  // causal: skip tiles past the diagonal
  for (int t = 0; t < n_tiles; ++t) {
    const int kv_base = t * BC;

    // ---- stage K (row major) and V (transposed) tiles into LDS as bf16 ----
    #pragma unroll
    for (int j = 0; j < 8; ++j) {
      const int idx = tid + 128 * j;   // 1024 float4 slots: 32 rows x 32 float4
      const int s   = idx >> 5;        // kv row in tile
      const int d4  = (idx & 31) * 4;  // head-dim col
      const size_t base =
          ((((size_t)b * kSk + kv_base + s) * 2 + 0) * kH + h) * kD + d4;
      const float4 kvec = *(const float4*)(kv + base);
      const float4 vvec = *(const float4*)(kv + base + (size_t)kH * kD); // V plane
      bf16_t* kd = &sK[s * KSTR + d4];
      kd[0] = (bf16_t)kvec.x; kd[1] = (bf16_t)kvec.y;
      kd[2] = (bf16_t)kvec.z; kd[3] = (bf16_t)kvec.w;
      sVt[(d4 + 0) * VSTR + s] = (bf16_t)vvec.x;
      sVt[(d4 + 1) * VSTR + s] = (bf16_t)vvec.y;
      sVt[(d4 + 2) * VSTR + s] = (bf16_t)vvec.z;
      sVt[(d4 + 3) * VSTR + s] = (bf16_t)vvec.w;
    }
    // prefetch next K tile (lowers to global_prefetch_b8)
    if (t + 1 < n_tiles) {
      const size_t nb =
          ((((size_t)b * kSk + kv_base + BC + (tid & 31)) * 2) * kH + h) * kD;
      __builtin_prefetch(kv + nb, 0, 1);
    }
    __syncthreads();

    // ---- S = Q * K^T ----
    // Group 1: all 16 ds_load_b128 for the 8 B-fragments.
    v16bf bk0[4], bk1[4];
    #pragma unroll
    for (int c = 0; c < 4; ++c) {
      const bf16_t* kb0 = &sK[ l16       * KSTR + c * 32 + half * 8];
      const bf16_t* kb1 = &sK[(l16 + 16) * KSTR + c * 32 + half * 8];
      v8bf x0 = *(const v8bf*)kb0, x1 = *(const v8bf*)(kb0 + 16);
      v8bf y0 = *(const v8bf*)kb1, y1 = *(const v8bf*)(kb1 + 16);
      #pragma unroll
      for (int e = 0; e < 8; ++e) {
        bk0[c][e] = x0[e]; bk0[c][e + 8] = x1[e];
        bk1[c][e] = y0[e]; bk1[c][e + 8] = y1[e];
      }
    }
    SCHED_FENCE();   // loads may not sink past this point
    // Group 2: the 8-WMMA chain (staggered dscnt waits, back-to-back issue).
    v8f c0 = vzero, c1 = vzero;
    #pragma unroll
    for (int c = 0; c < 4; ++c) {
      c0 = WMMA_BF16(aq[c], bk0[c], c0);
      c1 = WMMA_BF16(aq[c], bk1[c], c1);
    }
    SCHED_FENCE();

    // ---- online softmax over the 16x32 score block (rows i + 8*half) ----
    #pragma unroll
    for (int i = 0; i < 8; ++i) {
      const int rg  = q_base + wave * BR_WAVE + i + 8 * half;
      const int cg0 = kv_base + l16;
      const int cg1 = kv_base + 16 + l16;
      float s0 = (cg0 > rg) ? -1.0e30f : c0[i] * softmax_scale;
      float s1 = (cg1 > rg) ? -1.0e30f : c1[i] * softmax_scale;
      float mx = fmaxf(s0, s1);
      mx = fmaxf(mx, __shfl_xor(mx, 1, 16));
      mx = fmaxf(mx, __shfl_xor(mx, 2, 16));
      mx = fmaxf(mx, __shfl_xor(mx, 4, 16));
      mx = fmaxf(mx, __shfl_xor(mx, 8, 16));
      const float mnew = fmaxf(m_i[i], mx);
      const float corr = __expf(m_i[i] - mnew);
      const float e0 = __expf(s0 - mnew);
      const float e1 = __expf(s1 - mnew);
      float rs = e0 + e1;
      rs += __shfl_xor(rs, 1, 16);
      rs += __shfl_xor(rs, 2, 16);
      rs += __shfl_xor(rs, 4, 16);
      rs += __shfl_xor(rs, 8, 16);
      l_i[i] = l_i[i] * corr + rs;
      m_i[i] = mnew;
      #pragma unroll
      for (int nc = 0; nc < 8; ++nc) o[nc][i] *= corr;
      bf16_t* pr = &sP[(wave * BR_WAVE + i + 8 * half) * PSTR];
      pr[l16]      = (bf16_t)e0;
      pr[l16 + 16] = (bf16_t)e1;
    }

    // ---- O += P * V (same-wave LDS RAW: hardware DScnt ordering) ----
    {
      // Group 1: P fragment + all 8 V fragments (may overlap softmax above).
      const bf16_t* pr = &sP[(wave * BR_WAVE + l16) * PSTR + half * 8];
      v8bf plo = *(const v8bf*)pr;
      v8bf phi = *(const v8bf*)(pr + 16);
      v16bf ap;
      #pragma unroll
      for (int e = 0; e < 8; ++e) { ap[e] = plo[e]; ap[e + 8] = phi[e]; }

      v16bf bv[8];
      #pragma unroll
      for (int nc = 0; nc < 8; ++nc) {
        const bf16_t* vb = &sVt[(nc * 16 + l16) * VSTR + half * 8];
        v8bf bvl = *(const v8bf*)vb;
        v8bf bvh = *(const v8bf*)(vb + 16);
        #pragma unroll
        for (int e = 0; e < 8; ++e) { bv[nc][e] = bvl[e]; bv[nc][e + 8] = bvh[e]; }
      }
      SCHED_FENCE();   // loads may not sink past this point
      // Group 2: the 8-WMMA PV chain.
      #pragma unroll
      for (int nc = 0; nc < 8; ++nc)
        o[nc] = WMMA_BF16(ap, bv[nc], o[nc]);
    }
    __syncthreads();
  }

  // ---- normalize and write out (fp32) ----
  #pragma unroll
  for (int i = 0; i < 8; ++i) {
    const float inv = 1.0f / l_i[i];
    const int rg = q_base + wave * BR_WAVE + i + 8 * half;
    float* orow = out + (((size_t)b * kSq + rg) * kH + h) * kD;
    #pragma unroll
    for (int nc = 0; nc < 8; ++nc)
      orow[nc * 16 + l16] = o[nc][i] * inv;
  }
}

extern "C" void kernel_launch(void* const* d_in, const int* in_sizes, int n_in,
                              void* d_out, int out_size, void* d_ws, size_t ws_size,
                              hipStream_t stream) {
  (void)in_sizes; (void)n_in; (void)out_size; (void)d_ws; (void)ws_size;
  const float* q  = (const float*)d_in[0];
  const float* kv = (const float*)d_in[1];
  float* out = (float*)d_out;
  dim3 grid(kSq / BR, kH, kB);   // 32 x 16 x 2 blocks
  dim3 block(NWAVES * 32);       // 4 waves (wave32)
  hipLaunchKernelGGL(fa_fwd_bf16_wmma, grid, block, 0, stream, q, kv, out);
}
